// SNU_30159260352834
// MI455X (gfx1250) — matrix operand order repaired and checked by reading, not compiled, not run on hardware
//
#include <hip/hip_runtime.h>

// ---- CDNA5 WMMA vector types ----
typedef __attribute__((ext_vector_type(16))) __bf16 v16bf;
typedef __attribute__((ext_vector_type(8)))  __bf16 v8bf;
typedef __attribute__((ext_vector_type(8)))  float  v8f;

#define B_   128
#define I_   512
#define H_   512
#define T_   512
#define HB   128         // h-values per block
#define TC   32          // t-values per chunk (=> 128B contiguous output rows)
#define KC   32          // K per bf16 WMMA
#define NK   (I_ / KC)   // 16 K-steps
#define NCH  (T_ / TC)   // 16 chunks
#define WS   520         // Wl row stride (bf16): 1040B rows -> bank-spread
#define XS   520         // xl row stride (bf16)
#define XWS  33          // xw row stride (f32): 132B rows -> conflict-free
#define DECAY 0.8f
#define NTHREADS 256

// LDS layout (dynamic shared):
//  Wl  : HB*WS bf16 = 133120 B @ 0        (W^T tile, staged once)
//  xl0 : TC*XS bf16 =  33280 B @ 133120   (x chunk ping)
//  xl1 : TC*XS bf16 =  33280 B @ 166400   (x chunk pong)
//  xw  : HB*XWS f32 =  16896 B @ 199680   (GEMM result tile)
#define XL0_OFF 133120
#define XL1_OFF 166400
#define XW_OFF  199680
#define LDS_BYTES (XW_OFF + HB * XWS * 4)   // 216576 B < 320 KB/WGP

__global__ __launch_bounds__(NTHREADS)
void snu_fused(const float* __restrict__ x, const float* __restrict__ W,
               const float* __restrict__ bias, float* __restrict__ out)
{
    extern __shared__ char smem[];
    __bf16* Wl  = (__bf16*)smem;
    __bf16* xl0 = (__bf16*)(smem + XL0_OFF);
    __bf16* xl1 = (__bf16*)(smem + XL1_OFF);
    float*  xw  = (float*)(smem + XW_OFF);

    const int tid = threadIdx.x;
    const int lid = tid & 31;
    const int wv  = tid >> 5;          // wave 0..7, one 16-row h-tile each
    const int h0  = blockIdx.x * HB;   // h-block
    const int b   = blockIdx.y;        // batch

    const size_t xbase = (size_t)b * I_ * T_;
    const int st4 = (tid & 7) * 4;     // this thread's t offset within chunk
    const int si0 = tid >> 3;          // base i row (stride 32 per j)

    // Pipelined x staging: 16 float4 (global_load_b128) per thread per chunk,
    // issued one chunk ahead and held in VGPRs across the GEMM.
    float4 xreg[16];

    auto issue_loads = [&](int tc) {
        const float* gp = x + xbase + (size_t)si0 * T_ + tc * TC + st4;
        #pragma unroll
        for (int j = 0; j < 16; ++j) {
            const float* p = gp + (size_t)(32 * j) * T_;
            xreg[j] = *(const float4*)p;
            __builtin_prefetch(p + TC, 0, 1);   // L2 prefetch one more chunk out
        }
    };
    auto store_chunk = [&](__bf16* xl) {
        #pragma unroll
        for (int j = 0; j < 16; ++j) {
            const int i = si0 + 32 * j;
            xl[(st4 + 0) * XS + i] = (__bf16)xreg[j].x;
            xl[(st4 + 1) * XS + i] = (__bf16)xreg[j].y;
            xl[(st4 + 2) * XS + i] = (__bf16)xreg[j].z;
            xl[(st4 + 3) * XS + i] = (__bf16)xreg[j].w;
        }
    };

    issue_loads(0);   // chunk-0 x loads in flight during W staging

    // ---- stage W^T (bf16) into LDS: Wl[h_local][i] ----
    for (int idx = tid; idx < HB * I_; idx += NTHREADS) {
        int hl = idx & (HB - 1);
        int i  = idx >> 7;
        Wl[hl * WS + i] = (__bf16)W[(size_t)i * H_ + (h0 + hl)];
    }
    store_chunk(xl0);
    __syncthreads();

    // recurrence state: threads 0..HB-1 each own one h
    float hs = 0.f, ys = 0.f, bv = 0.f;
    if (tid < HB) bv = bias[h0 + tid];

    // per-lane WMMA fragment geometry (ISA 7.12.2 layouts)
    const int hrow  = wv * 16 + (lid & 15);   // A: row M
    const int halfA = (lid < 16) ? 0 : 8;     // A: K blocks {0-7,16-23}|{8-15,24-31}
    const int halfB = (lid < 16) ? 0 : 8;     // B: K blocks {0-7,16-23}|{8-15,24-31}
    const int ncol  = lid & 15;               // B/C: column N

    auto ldA = [&](int kb, v16bf& a) {
        v8bf lo = *(const v8bf*)&Wl[hrow * WS + kb + halfA];
        v8bf hi = *(const v8bf*)&Wl[hrow * WS + kb + 16 + halfA];
        a = __builtin_shufflevector(lo, hi, 0,1,2,3,4,5,6,7,8,9,10,11,12,13,14,15);
    };
    auto ldB = [&](const __bf16* base, int kb, int nt, v16bf& bb) {
        const __bf16* rp = &base[(nt * 16 + ncol) * XS + kb + halfB];
        v8bf lo = *(const v8bf*)rp;
        v8bf hi = *(const v8bf*)(rp + 16);
        bb = __builtin_shufflevector(lo, hi, 0,1,2,3,4,5,6,7,8,9,10,11,12,13,14,15);
    };

    for (int tc = 0; tc < NCH; ++tc) {
        const __bf16* xl = (tc & 1) ? xl1 : xl0;
        __bf16* xln      = (tc & 1) ? xl0 : xl1;

        if (tc + 1 < NCH) issue_loads(tc + 1);   // HBM streams during GEMM

        // ---- GEMM: 16h x 32t f32 tile per wave over K=512 ----
        // Ping-pong fragment sets: no register rotation -> no movs, no
        // WMMA->VALU WAR hazard NOPs; partial s_wait_dscnt overlap retained.
        v8f c0 = {0.f,0.f,0.f,0.f,0.f,0.f,0.f,0.f};
        v8f c1 = {0.f,0.f,0.f,0.f,0.f,0.f,0.f,0.f};
        v16bf a0, b00, b10, a1, b01, b11;
        ldA(0, a0); ldB(xl, 0, 0, b00); ldB(xl, 0, 1, b10);
        #pragma unroll 4
        for (int kk = 0; kk < NK; kk += 2) {
            {   // prefetch odd step's fragments (set 1)
                const int kb = (kk + 1) * KC;
                ldA(kb, a1); ldB(xl, kb, 0, b01); ldB(xl, kb, 1, b11);
            }
            c0 = __builtin_amdgcn_wmma_f32_16x16x32_bf16(
                     false, a0, false, b00, (short)0, c0, false, false);
            c1 = __builtin_amdgcn_wmma_f32_16x16x32_bf16(
                     false, a0, false, b10, (short)0, c1, false, false);
            if (kk + 2 < NK) {   // prefetch next even step's fragments (set 0)
                const int kb = (kk + 2) * KC;
                ldA(kb, a0); ldB(xl, kb, 0, b00); ldB(xl, kb, 1, b10);
            }
            c0 = __builtin_amdgcn_wmma_f32_16x16x32_bf16(
                     false, a1, false, b01, (short)0, c0, false, false);
            c1 = __builtin_amdgcn_wmma_f32_16x16x32_bf16(
                     false, a1, false, b11, (short)0, c1, false, false);
        }

        // ---- spill C tiles to LDS (C layout: VGPR r -> M = r + 8*(lane>=16)) ----
        {
            const int hl = wv * 16 + 8 * (lid >> 4);
            #pragma unroll
            for (int r = 0; r < 8; ++r) {
                xw[(hl + r) * XWS + ncol]      = c0[r];
                xw[(hl + r) * XWS + 16 + ncol] = c1[r];
            }
        }
        // convert + stage next chunk into the other buffer (loadcnt wait lands here)
        if (tc + 1 < NCH) store_chunk(xln);
        __syncthreads();

        // ---- sequential SNU recurrence over the 32 t-steps ----
        if (tid < HB) {
            float4 o[TC / 4];
            const float* row = &xw[tid * XWS];
            #pragma unroll
            for (int n = 0; n < TC; ++n) {
                float v = row[n];
                hs = fmaxf(v + DECAY * hs * (1.0f - ys), 0.0f);
                ys = (hs + bv > 0.0f) ? 1.0f : 0.0f;
                ((float*)o)[n] = ys;
            }
            float4* op = (float4*)(out + ((size_t)b * H_ + (h0 + tid)) * T_ + tc * TC);
            #pragma unroll
            for (int q = 0; q < TC / 4; ++q) op[q] = o[q];
        }
        __syncthreads();
    }
}

extern "C" void kernel_launch(void* const* d_in, const int* in_sizes, int n_in,
                              void* d_out, int out_size, void* d_ws, size_t ws_size,
                              hipStream_t stream) {
    (void)in_sizes; (void)n_in; (void)d_ws; (void)ws_size; (void)out_size;
    const float* x    = (const float*)d_in[0];   // (B, I, T)
    const float* W    = (const float*)d_in[1];   // (I, H)
    const float* bias = (const float*)d_in[2];   // (1, H)
    float* out        = (float*)d_out;           // (B, H, T)

    dim3 grid(H_ / HB, B_);                      // 4 x 128 = 512 blocks
    snu_fused<<<grid, NTHREADS, LDS_BYTES, stream>>>(x, W, bias, out);
}
